// Tmp_489626271910
// MI455X (gfx1250) — compile-verified, matrix operand order checked by reading, not run on hardware
//
#include <hip/hip_runtime.h>

#define CDIM 128
#define NDIM 32768
#define LDP  144                 // padded LDS row (bf16 elems): 288B rows = 64 data dw + 8 pad dw
#define LOG2E 1.4426950408889634f
#define CHUNKS 16
#define NCHUNK (NDIM / CHUNKS)   // 2048

typedef __bf16 bf16;
typedef __attribute__((ext_vector_type(16))) bf16  v16bf;
typedef __attribute__((ext_vector_type(8)))  bf16  v8bf;
typedef __attribute__((ext_vector_type(4)))  bf16  v4bf;
typedef __attribute__((ext_vector_type(2)))  bf16  v2bf;
typedef __attribute__((ext_vector_type(8)))  float v8f;
typedef unsigned v4u __attribute__((ext_vector_type(4)));
typedef int      v4i __attribute__((ext_vector_type(4)));
typedef int      v8i __attribute__((ext_vector_type(8)));

// A(16x32 bf16) and B(32x16 bf16) fragments share the same address function:
// lane&15 -> row (A: M, B: column), lane>>4 selects k half; 16 contiguous bf16.
static __device__ __forceinline__ v16bf ldfrag(const bf16* base, int row0, int k0, int lane) {
  const bf16* p = base + (size_t)(row0 + (lane & 15)) * LDP + k0 + ((lane >> 4) << 4);
  return *(const v16bf*)p;
}

static __device__ __forceinline__ v8f wmma_bf16(v16bf a, v16bf b, v8f c) {
  return __builtin_amdgcn_wmma_f32_16x16x32_bf16(false, a, false, b, (short)0, c, false, false);
}

// monotone float<->uint mapping for atomicMax on floats
static __device__ __forceinline__ unsigned fenc(float f) {
  unsigned u = __float_as_uint(f);
  return (u & 0x80000000u) ? ~u : (u | 0x80000000u);
}
static __device__ __forceinline__ float fdec(unsigned u) {
  return (u & 0x80000000u) ? __uint_as_float(u & 0x7FFFFFFFu) : __uint_as_float(~u);
}

// TDM: DMA a 128x128 tile of 2-byte elements (row length dim0, row stride stride0,
// dim1 rows total in tensor) from global into LDS with 64dw+8dw row padding (-> LDP rows).
static __device__ __forceinline__ void tdm_load_tile(const void* gsrc, void* ldst,
                                                     unsigned dim0, unsigned dim1,
                                                     unsigned stride0) {
  unsigned long long ga = (unsigned long long)(size_t)gsrc;
  unsigned lds_off = (unsigned)(size_t)ldst;   // generic LDS addr low 32 bits = LDS offset
  v4u g0;
  g0[0] = 1u;                                          // count=1, user descriptor
  g0[1] = lds_off;                                     // LDS byte address
  g0[2] = (unsigned)(ga & 0xFFFFFFFFu);                // global_addr[31:0]
  g0[3] = (unsigned)((ga >> 32) & 0x01FFFFFFu) | 0x80000000u;  // addr[56:32] | type=2
  v8i g1;
  g1[0] = (int)((1u << 16) | (1u << 20) | (5u << 22) | (7u << 25)); // 2B elems, pad_en, 64dw interval, 8dw pad
  g1[1] = (int)((dim0 & 0xFFFFu) << 16);               // tensor_dim0[15:0]
  g1[2] = (int)(((dim0 >> 16) & 0xFFFFu) | ((dim1 & 0xFFFFu) << 16)); // dim0 hi | dim1 lo
  g1[3] = (int)(((dim1 >> 16) & 0xFFFFu) | (128u << 16));            // dim1 hi | tile_dim0=128
  g1[4] = (int)128u;                                   // tile_dim1=128, tile_dim2=0
  g1[5] = (int)stride0;                                // tensor_dim0_stride[31:0]
  g1[6] = 0;                                           // stride0 hi | stride1 lo
  g1[7] = 0;
  v4i gz4 = {0, 0, 0, 0};
  v8i gz8 = {0, 0, 0, 0, 0, 0, 0, 0};
  __builtin_amdgcn_tensor_load_to_lds(g0, g1, gz4, gz4, gz8, 0);
}

__global__ __launch_bounds__(256) void init_kernel(float* __restrict__ Wctx,
                                                   float* __restrict__ ksum,
                                                   unsigned* __restrict__ kmaxb) {
  int i = blockIdx.x * 256 + threadIdx.x;
  if (i < 4 * CDIM * CDIM) Wctx[i] = 0.0f;
  if (i < 4 * CDIM) { ksum[i] = 0.0f; kmaxb[i] = 0x007FFFFFu; /* fenc(-inf) */ }
}

// -------- Kernel 1: fused projections, Q-softmax, K row-max -----------------
// Output layouts (per combo = s*2+b): Kt/Vt = (C, N) bf16, Qn = (N, C) bf16.
__global__ __launch_bounds__(256) void proj_kernel(
    const float* __restrict__ x0, const float* __restrict__ x1,
    const float* __restrict__ Wk0, const float* __restrict__ Wk1,
    const float* __restrict__ Wq0, const float* __restrict__ Wq1,
    const float* __restrict__ Wv0, const float* __restrict__ Wv1,
    bf16* __restrict__ Kt, bf16* __restrict__ Vt, bf16* __restrict__ Qn,
    unsigned* __restrict__ kmaxb)
{
  __shared__ bf16 Alds[CDIM][LDP] __attribute__((aligned(32)));  // weights [c_out][c]
  __shared__ bf16 Blds[CDIM][LDP] __attribute__((aligned(32)));  // x tile  [n][c]
  __shared__ float Slds[CDIM][132];                              // Q softmax scratch

  const int t = threadIdx.x, lane = t & 31, wave = t >> 5;
  const int combo = blockIdx.y;                 // s*2 + b
  const int s = combo >> 1, b = combo & 1;
  const int n0 = blockIdx.x * 128;
  const float* xp = (s ? x1 : x0) + (size_t)b * CDIM * NDIM;
  const float* Wk = s ? Wk1 : Wk0;
  const float* Wq = s ? Wq1 : Wq0;
  const float* Wv = s ? Wv1 : Wv0;
  const size_t cb = (size_t)combo * CDIM * NDIM;

  // load x tile (128 c x 128 n), bf16-convert, transpose into Blds[n][c]
  for (int e = t; e < CDIM * 32; e += 256) {
    int c = e >> 5, q = e & 31;
    float4 v = *(const float4*)(xp + (size_t)c * NDIM + n0 + q * 4);
    Blds[q * 4 + 0][c] = (bf16)v.x;
    Blds[q * 4 + 1][c] = (bf16)v.y;
    Blds[q * 4 + 2][c] = (bf16)v.z;
    Blds[q * 4 + 3][c] = (bf16)v.w;
  }

  auto load_w = [&](const float* Wp) {
    for (int e = t; e < CDIM * 32; e += 256) {
      int co = e >> 5, q = e & 31;
      float4 v = *(const float4*)(Wp + co * CDIM + q * 4);
      v4bf pk; pk[0] = (bf16)v.x; pk[1] = (bf16)v.y; pk[2] = (bf16)v.z; pk[3] = (bf16)v.w;
      *(v4bf*)&Alds[co][q * 4] = pk;
    }
  };

  auto run_gemm = [&](v8f* acc) {
#pragma unroll
    for (int ks = 0; ks < 4; ks++) {
      v16bf a = ldfrag(&Alds[0][0], wave * 16, ks * 32, lane);
      v16bf bfr[8];
#pragma unroll
      for (int nt = 0; nt < 8; nt++) bfr[nt] = ldfrag(&Blds[0][0], nt * 16, ks * 32, lane);
#pragma unroll
      for (int nt = 0; nt < 8; nt++) acc[nt] = wmma_bf16(a, bfr[nt], acc[nt]);
    }
  };
  const v8f vz = {0.f, 0.f, 0.f, 0.f, 0.f, 0.f, 0.f, 0.f};

  // ---- V projection ----
  load_w(Wv);
  __syncthreads();
  {
    v8f acc[8];
#pragma unroll
    for (int i = 0; i < 8; i++) acc[i] = vz;
    run_gemm(acc);
#pragma unroll
    for (int nt = 0; nt < 8; nt++)
#pragma unroll
      for (int r = 0; r < 8; r++) {
        int co = wave * 16 + r + ((lane >> 4) << 3);
        int n = n0 + nt * 16 + (lane & 15);
        Vt[cb + (size_t)co * NDIM + n] = (bf16)acc[nt][r];
      }
  }
  __syncthreads();

  // ---- K projection (store logits + per-row running max over N) ----
  load_w(Wk);
  __syncthreads();
  {
    v8f acc[8];
#pragma unroll
    for (int i = 0; i < 8; i++) acc[i] = vz;
    run_gemm(acc);
#pragma unroll
    for (int nt = 0; nt < 8; nt++)
#pragma unroll
      for (int r = 0; r < 8; r++) {
        int co = wave * 16 + r + ((lane >> 4) << 3);
        int n = n0 + nt * 16 + (lane & 15);
        Kt[cb + (size_t)co * NDIM + n] = (bf16)acc[nt][r];
      }
#pragma unroll
    for (int r = 0; r < 8; r++) {
      float m = acc[0][r];
#pragma unroll
      for (int nt = 1; nt < 8; nt++) m = fmaxf(m, acc[nt][r]);
      m = fmaxf(m, __shfl_xor(m, 1));
      m = fmaxf(m, __shfl_xor(m, 2));
      m = fmaxf(m, __shfl_xor(m, 4));
      m = fmaxf(m, __shfl_xor(m, 8));
      if ((lane & 15) == 0) {
        int co = wave * 16 + r + ((lane >> 4) << 3);
        atomicMax(&kmaxb[combo * CDIM + co], fenc(m));
      }
    }
  }
  __syncthreads();

  // ---- Q projection + softmax over C (per n) ----
  load_w(Wq);
  __syncthreads();
  {
    v8f acc[8];
#pragma unroll
    for (int i = 0; i < 8; i++) acc[i] = vz;
    run_gemm(acc);
#pragma unroll
    for (int nt = 0; nt < 8; nt++)
#pragma unroll
      for (int r = 0; r < 8; r++) {
        int row = wave * 16 + r + ((lane >> 4) << 3);
        int col = nt * 16 + (lane & 15);
        Slds[row][col] = acc[nt][r];
      }
    __syncthreads();
    if (t < CDIM) {
      int j = t;
      float m = -3.4e38f;
      for (int c = 0; c < CDIM; c++) m = fmaxf(m, Slds[c][j]);
      float ssum = 0.0f;
      for (int c = 0; c < CDIM; c++) {
        float p = exp2f((Slds[c][j] - m) * LOG2E);
        Slds[c][j] = p;
        ssum += p;
      }
      float inv = 1.0f / ssum;
      bf16* qrow = Qn + cb + (size_t)(n0 + j) * CDIM;
      for (int c = 0; c < CDIM; c += 2) {
        v2bf pk;
        pk[0] = (bf16)(Slds[c][j] * inv);
        pk[1] = (bf16)(Slds[c + 1][j] * inv);
        *(v2bf*)&qrow[c] = pk;
      }
    }
  }
}

// -------- Kernel 2: context GEMM  W[c,d] += sum_n exp(K[c,n]-max_c)*V[d,n] ---
// V tile staged by the Tensor Data Mover (overlaps with exp() on the K tile).
__global__ __launch_bounds__(256) void ctx_kernel(
    const bf16* __restrict__ Kt, const bf16* __restrict__ Vt,
    const unsigned* __restrict__ kmaxb,
    float* __restrict__ Wctx, float* __restrict__ ksum)
{
  __shared__ bf16 Elds[CDIM][LDP] __attribute__((aligned(32)));
  __shared__ bf16 Vlds[CDIM][LDP] __attribute__((aligned(32)));
  __shared__ float kmaxL[CDIM];
  __shared__ float ksumL[CDIM];

  const int t = threadIdx.x, lane = t & 31, wave = t >> 5;
  const int combo = blockIdx.y;
  const size_t cb = (size_t)combo * CDIM * NDIM;
  const int nbase = blockIdx.x * NCHUNK;

  if (t < CDIM) { kmaxL[t] = fdec(kmaxb[combo * CDIM + t]); ksumL[t] = 0.0f; }
  __syncthreads();

  const v8f vz = {0.f, 0.f, 0.f, 0.f, 0.f, 0.f, 0.f, 0.f};
  v8f acc[8];
#pragma unroll
  for (int i = 0; i < 8; i++) acc[i] = vz;

  for (int nb = nbase; nb < nbase + NCHUNK; nb += 128) {
    // async DMA: V tile (128 c x 128 n) -> Vlds with padded rows
    if (t < 32) tdm_load_tile(Vt + cb + nb, &Vlds[0][0], NDIM, CDIM, NDIM);

    // meanwhile: stage exp(K - max) tile with VALU, accumulate row sums
    for (int e = t; e < CDIM * 16; e += 256) {
      int c = e >> 4, seg = e & 15;
      const bf16* kp = Kt + cb + (size_t)c * NDIM + nb + seg * 8;
      v8bf kv = *(const v8bf*)kp;
      __builtin_prefetch(kp + 128, 0, 0);   // next chunk -> global_prefetch_b8
      float km = kmaxL[c];
      float part = 0.0f;
      v8bf ev;
#pragma unroll
      for (int j = 0; j < 8; j++) {
        float p = exp2f(((float)kv[j] - km) * LOG2E);
        part += p;
        ev[j] = (bf16)p;
      }
      *(v8bf*)&Elds[c][seg * 8] = ev;
      atomicAdd(&ksumL[c], part);
    }
    if (t < 32) __builtin_amdgcn_s_wait_tensorcnt((short)0);
    __syncthreads();
#pragma unroll
    for (int ks = 0; ks < 4; ks++) {
      v16bf a = ldfrag(&Elds[0][0], wave * 16, ks * 32, lane);
      v16bf bfr[8];
#pragma unroll
      for (int dt = 0; dt < 8; dt++) bfr[dt] = ldfrag(&Vlds[0][0], dt * 16, ks * 32, lane);
#pragma unroll
      for (int dt = 0; dt < 8; dt++) acc[dt] = wmma_bf16(a, bfr[dt], acc[dt]);
    }
    __syncthreads();
  }

  float* Wp = Wctx + combo * CDIM * CDIM;
#pragma unroll
  for (int dt = 0; dt < 8; dt++)
#pragma unroll
    for (int r = 0; r < 8; r++) {
      int c = wave * 16 + r + ((lane >> 4) << 3);
      int d = dt * 16 + (lane & 15);
      atomicAdd(&Wp[c * CDIM + d], acc[dt][r]);
    }
  __syncthreads();
  if (t < CDIM) atomicAdd(&ksum[combo * CDIM + t], ksumL[t]);
}

// -------- Kernel 3: Out_s[d,n] = sum_c (W_{1-s}[c,d]/ksum_c) * Q_s[n,c] -----
// Q tile staged by the Tensor Data Mover.
__global__ __launch_bounds__(256) void out_kernel(
    const bf16* __restrict__ Qn, const float* __restrict__ Wctx,
    const float* __restrict__ ksum, float* __restrict__ out)
{
  __shared__ bf16 Alds[CDIM][LDP] __attribute__((aligned(32)));  // [d][c]
  __shared__ bf16 Qlds[CDIM][LDP] __attribute__((aligned(32)));  // [n][c]
  __shared__ float ksumL[CDIM];

  const int t = threadIdx.x, lane = t & 31, wave = t >> 5;
  const int combo = blockIdx.y;
  const int s = combo >> 1, b = combo & 1;
  const int other = ((1 - s) << 1) | b;
  const int n0 = blockIdx.x * 128;
  const bf16* Qb = Qn + (size_t)combo * CDIM * NDIM;

  // async DMA: Q tile (128 n x 128 c) -> Qlds with padded rows
  if (t < 32) tdm_load_tile(Qb + (size_t)n0 * CDIM, &Qlds[0][0], CDIM, NDIM, CDIM);

  if (t < CDIM) ksumL[t] = ksum[other * CDIM + t];
  __syncthreads();

  const float* Wp = Wctx + other * CDIM * CDIM;
  for (int e = t; e < CDIM * 16; e += 256) {
    int d = e >> 4, seg = e & 15;
#pragma unroll
    for (int j = 0; j < 8; j++) {
      int c = seg * 8 + j;
      Alds[d][c] = (bf16)(Wp[c * CDIM + d] / ksumL[c]);
    }
  }
  if (t < 32) __builtin_amdgcn_s_wait_tensorcnt((short)0);
  __syncthreads();

  const v8f vz = {0.f, 0.f, 0.f, 0.f, 0.f, 0.f, 0.f, 0.f};
  v8f acc[8];
#pragma unroll
  for (int i = 0; i < 8; i++) acc[i] = vz;
#pragma unroll
  for (int ks = 0; ks < 4; ks++) {
    v16bf a = ldfrag(&Alds[0][0], wave * 16, ks * 32, lane);
    v16bf bfr[8];
#pragma unroll
    for (int nt = 0; nt < 8; nt++) bfr[nt] = ldfrag(&Qlds[0][0], nt * 16, ks * 32, lane);
#pragma unroll
    for (int nt = 0; nt < 8; nt++) acc[nt] = wmma_bf16(a, bfr[nt], acc[nt]);
  }

  float* ob = out + (size_t)combo * CDIM * NDIM;
#pragma unroll
  for (int nt = 0; nt < 8; nt++)
#pragma unroll
    for (int r = 0; r < 8; r++) {
      int d = wave * 16 + r + ((lane >> 4) << 3);
      int n = n0 + nt * 16 + (lane & 15);
      ob[(size_t)d * NDIM + n] = acc[nt][r];
    }
}

extern "C" void kernel_launch(void* const* d_in, const int* in_sizes, int n_in,
                              void* d_out, int out_size, void* d_ws, size_t ws_size,
                              hipStream_t stream) {
  (void)in_sizes; (void)n_in; (void)out_size; (void)ws_size;
  const float* x0  = (const float*)d_in[0];
  const float* x1  = (const float*)d_in[1];
  const float* Wk0 = (const float*)d_in[2];
  const float* Wk1 = (const float*)d_in[3];
  const float* Wq0 = (const float*)d_in[4];
  const float* Wq1 = (const float*)d_in[5];
  const float* Wv0 = (const float*)d_in[6];
  const float* Wv1 = (const float*)d_in[7];
  float* out = (float*)d_out;

  const size_t PLANE = (size_t)4 * CDIM * NDIM;   // elems per bf16 array (all combos)
  bf16* Kt = (bf16*)d_ws;
  bf16* Vt = Kt + PLANE;
  bf16* Qn = Vt + PLANE;
  float* Wctx = (float*)(Qn + PLANE);
  float* ksum = Wctx + 4 * CDIM * CDIM;
  unsigned* kmaxb = (unsigned*)(ksum + 4 * CDIM);

  init_kernel<<<dim3((4 * CDIM * CDIM + 255) / 256), 256, 0, stream>>>(Wctx, ksum, kmaxb);
  proj_kernel<<<dim3(NDIM / 128, 4), 256, 0, stream>>>(
      x0, x1, Wk0, Wk1, Wq0, Wq1, Wv0, Wv1, Kt, Vt, Qn, kmaxb);
  ctx_kernel<<<dim3(CHUNKS, 4), 256, 0, stream>>>(Kt, Vt, kmaxb, Wctx, ksum);
  out_kernel<<<dim3(NDIM / 128, 4), 256, 0, stream>>>(Qn, Wctx, ksum, out);
}
